// Ansatz_fb_72705206387209
// MI455X (gfx1250) — matrix operand order repaired and checked by reading, not compiled, not run on hardware
//
#include <hip/hip_runtime.h>
#include <math.h>

#define NE 512
#define NU 256
#define NA 16

typedef __attribute__((ext_vector_type(16))) __bf16 v16bf;
typedef __attribute__((ext_vector_type(8)))  float  v8f;

__device__ inline v8f wmma_bf16(v16bf a, v16bf b, v8f c) {
  return __builtin_amdgcn_wmma_f32_16x16x32_bf16(false, a, false, b, (short)0, c, false, false);
}

// A fragment: 16x32 (MxK) bf16, converted from f32 rows. Lane l(0-15) <-> row m0+l.
// VGPR p: K = (p<4 ? hi*8+2p : 16+hi*8+2(p-4)) per ISA 16-bit A layout.
__device__ inline v16bf load_a_frag(const float* A, int lda, int m0, int kbase, int K) {
  int lane = threadIdx.x & 31;
  int l = lane & 15, hi = lane >> 4;
  const float* row = A + (size_t)(m0 + l) * lda;
  v16bf af;
#pragma unroll
  for (int p = 0; p < 8; ++p) {
    int k = kbase + (p < 4 ? hi * 8 + 2 * p : 16 + hi * 8 + 2 * (p - 4));
    float f0 = (k     < K) ? row[k]     : 0.0f;
    float f1 = (k + 1 < K) ? row[k + 1] : 0.0f;
    af[2 * p]     = (__bf16)f0;
    af[2 * p + 1] = (__bf16)f1;
  }
  return af;
}

// B fragment: 32x16 (KxN) bf16 from f32 row-major W (ldw = N). Lane l <-> col n0+l,
// VGPR p: K = (p<4 ? hi*8+2p : 16+hi*8+2(p-4)), 2 K-values packed per VGPR.
__device__ inline v16bf load_b_frag(const float* W, int ldw, int kbase, int n0, int K) {
  int lane = threadIdx.x & 31;
  int l = lane & 15, hi = lane >> 4;
  int n = n0 + l;
  v16bf bf;
#pragma unroll
  for (int p = 0; p < 8; ++p) {
    int k = kbase + (p < 4 ? hi * 8 + 2 * p : 16 + hi * 8 + 2 * (p - 4));
    float f0 = (k     < K) ? W[(size_t)k * ldw + n]       : 0.0f;
    float f1 = (k + 1 < K) ? W[(size_t)(k + 1) * ldw + n] : 0.0f;
    bf[2 * p]     = (__bf16)f0;
    bf[2 * p + 1] = (__bf16)f1;
  }
  return bf;
}

// ---- prep: ra features, s_v0 (512x64), exp-envelope sums -------------------
__global__ void prep_kernel(const float* __restrict__ r, const float* __restrict__ a,
                            float* __restrict__ s_v0, float* __restrict__ e_fac) {
  int i = blockIdx.x * blockDim.x + threadIdx.x;
  if (i >= NE) return;
  float rx = r[3 * i], ry = r[3 * i + 1], rz = r[3 * i + 2];
  float es = 0.f;
#pragma unroll 4
  for (int q = 0; q < NA; ++q) {
    float dx = rx - a[3 * q], dy = ry - a[3 * q + 1], dz = rz - a[3 * q + 2];
    float len = sqrtf(dx * dx + dy * dy + dz * dz);
    s_v0[i * 64 + 4 * q + 0] = dx;
    s_v0[i * 64 + 4 * q + 1] = dy;
    s_v0[i * 64 + 4 * q + 2] = dz;
    s_v0[i * 64 + 4 * q + 3] = len;
    es += expf(-len);
  }
  e_fac[i] = es;
}

// ---- column means of s over spin halves ------------------------------------
__global__ void smean_kernel(const float* __restrict__ S, int d,
                             float* __restrict__ su, float* __restrict__ sd) {
  int c = blockIdx.x * blockDim.x + threadIdx.x;
  if (c >= d) return;
  float s1 = 0.f, s2 = 0.f;
  for (int i = 0; i < NU; ++i)   s1 += S[(size_t)i * d + c];
  for (int i = NU; i < NE; ++i)  s2 += S[(size_t)i * d + c];
  su[c] = s1 * (1.f / NU);
  sd[c] = s2 * (1.f / NU);
}

// ---- initial p4 means over i (p4 never materialized) -----------------------
__global__ void p4mean_kernel(const float* __restrict__ r,
                              float* __restrict__ pu, float* __restrict__ pd) {
  __shared__ float acc[8];
  if (threadIdx.x < 8) acc[threadIdx.x] = 0.f;
  __syncthreads();
  int j = blockIdx.x;
  float rj0 = r[3 * j], rj1 = r[3 * j + 1], rj2 = r[3 * j + 2];
#pragma unroll
  for (int h = 0; h < 2; ++h) {
    int i = threadIdx.x + h * NU;
    float dx = rj0 - r[3 * i], dy = rj1 - r[3 * i + 1], dz = rj2 - r[3 * i + 2];
    float ex = dx, ey = dy, ez = dz;
    if (i == j) { ex += 1.f; ey += 1.f; ez += 1.f; }
    float len = sqrtf(ex * ex + ey * ey + ez * ez);
    atomicAdd(&acc[h * 4 + 0], dx);
    atomicAdd(&acc[h * 4 + 1], dy);
    atomicAdd(&acc[h * 4 + 2], dz);
    atomicAdd(&acc[h * 4 + 3], len);
  }
  __syncthreads();
  if (threadIdx.x < 4)      pu[j * 4 + threadIdx.x]       = acc[threadIdx.x] * (1.f / NU);
  else if (threadIdx.x < 8) pd[j * 4 + threadIdx.x - 4]   = acc[threadIdx.x] * (1.f / NU);
}

// ---- layer-0 p transform: P[pair] = tanh(p4 @ W0 + b0), rank-4 -> VALU -----
__global__ void p0_kernel(const float* __restrict__ r, const float* __restrict__ W0,
                          const float* __restrict__ b0, float* __restrict__ P) {
  __shared__ float sW[4 * 64];
  __shared__ float sb[64];
  for (int idx = threadIdx.x; idx < 256; idx += blockDim.x) sW[idx] = W0[idx];
  if (threadIdx.x < 64) sb[threadIdx.x] = b0[threadIdx.x];
  __syncthreads();
  size_t pair = (size_t)blockIdx.x * 4 + (threadIdx.x >> 6);
  int c = threadIdx.x & 63;
  int i = (int)(pair >> 9), j = (int)(pair & 511);
  float dx = r[3 * j] - r[3 * i], dy = r[3 * j + 1] - r[3 * i + 1], dz = r[3 * j + 2] - r[3 * i + 2];
  float ex = dx, ey = dy, ez = dz;
  if (i == j) { ex += 1.f; ey += 1.f; ez += 1.f; }
  float len = sqrtf(ex * ex + ey * ey + ez * ez);
  float v = sb[c] + dx * sW[c] + dy * sW[64 + c] + dz * sW[128 + c] + len * sW[192 + c];
  P[pair * 64 + c] = tanhf(v);
}

// ---- p means over i for current P (512x512x64) -----------------------------
__global__ void pmean_kernel(const float* __restrict__ P,
                             float* __restrict__ pu, float* __restrict__ pd) {
  int j = blockIdx.x, c = threadIdx.x;  // blockDim = 64
  float su = 0.f, sd = 0.f;
  for (int i = 0; i < NU; ++i)  su += P[((size_t)i * NE + j) * 64 + c];
  for (int i = NU; i < NE; ++i) sd += P[((size_t)i * NE + j) * 64 + c];
  pu[j * 64 + c] = su * (1.f / NU);
  pd[j * 64 + c] = sd * (1.f / NU);
}

// ---- residual p layer: P <- tanh(P @ W(64x64) + b) + P, via bf16 WMMA ------
// block = 256 threads = 8 waves; each wave owns a 16-row strip x 64 cols.
__global__ __launch_bounds__(256) void pgemm_kernel(float* __restrict__ P,
                                                    const float* __restrict__ W,
                                                    const float* __restrict__ b) {
  __shared__ float sW[64 * 64];
  __shared__ float sB[64];
  for (int idx = threadIdx.x; idx < 64 * 64; idx += blockDim.x) sW[idx] = W[idx];
  if (threadIdx.x < 64) sB[threadIdx.x] = b[threadIdx.x];
  __syncthreads();
  int wave = threadIdx.x >> 5;
  size_t m0 = ((size_t)blockIdx.x * 8 + wave) * 16;  // row in 262144
  float* base = P + m0 * 64;

  v16bf a0 = load_a_frag(base, 64, 0, 0, 64);
  v16bf a1 = load_a_frag(base, 64, 0, 32, 64);
  v8f c0 = {}, c1 = {}, c2 = {}, c3 = {};
  c0 = wmma_bf16(a0, load_b_frag(sW, 64, 0, 0, 64), c0);
  c0 = wmma_bf16(a1, load_b_frag(sW, 64, 32, 0, 64), c0);
  c1 = wmma_bf16(a0, load_b_frag(sW, 64, 0, 16, 64), c1);
  c1 = wmma_bf16(a1, load_b_frag(sW, 64, 32, 16, 64), c1);
  c2 = wmma_bf16(a0, load_b_frag(sW, 64, 0, 32, 64), c2);
  c2 = wmma_bf16(a1, load_b_frag(sW, 64, 32, 32, 64), c2);
  c3 = wmma_bf16(a0, load_b_frag(sW, 64, 0, 48, 64), c3);
  c3 = wmma_bf16(a1, load_b_frag(sW, 64, 32, 48, 64), c3);

  int lane = threadIdx.x & 31;
  int l = lane & 15, hi = lane >> 4;
  v8f cc[4] = {c0, c1, c2, c3};
#pragma unroll
  for (int nt = 0; nt < 4; ++nt) {
#pragma unroll
    for (int g = 0; g < 8; ++g) {
      int m = g + hi * 8;
      int n = nt * 16 + l;
      size_t idx = (size_t)m * 64 + n;
      base[idx] = tanhf(cc[nt][g] + sB[n]) + base[idx];
    }
  }
}

// ---- build s_blk = [su_m | sd_m | pu_m[e] | pd_m[e] | s_v[e]] --------------
__global__ void sblk_kernel(const float* __restrict__ S, const float* __restrict__ su,
                            const float* __restrict__ sd, const float* __restrict__ pu,
                            const float* __restrict__ pd, int ds, int dp,
                            float* __restrict__ blk) {
  int e = blockIdx.x;
  int Din = 3 * ds + 2 * dp;
  for (int c = threadIdx.x; c < Din; c += blockDim.x) {
    float v;
    if (c < ds)                 v = su[c];
    else if (c < 2 * ds)        v = sd[c - ds];
    else if (c < 2 * ds + dp)   v = pu[e * dp + (c - 2 * ds)];
    else if (c < 2 * ds + 2 * dp) v = pd[e * dp + (c - 2 * ds - dp)];
    else                        v = S[(size_t)e * ds + (c - 2 * ds - 2 * dp)];
    blk[(size_t)e * Din + c] = v;
  }
}

// ---- generic s-path GEMM: out = act(A(MxK)@W(KxN)+b) [*rowscale] [+res] ----
// one wave per block; wave tile = 16 rows x 32 cols; grid = (M/16, N/32).
__global__ void sgemm_kernel(const float* __restrict__ A, int K,
                             const float* __restrict__ W, const float* __restrict__ bias,
                             float* __restrict__ out, int N,
                             const float* __restrict__ residual,
                             const float* __restrict__ rowscale, int do_tanh) {
  int m0 = blockIdx.x * 16;
  int n0 = blockIdx.y * 32;
  v8f c0 = {}, c1 = {};
  int nk = (K + 31) / 32;
  for (int kc = 0; kc < nk; ++kc) {
    v16bf af = load_a_frag(A, K, m0, kc * 32, K);
    c0 = wmma_bf16(af, load_b_frag(W, N, kc * 32, n0, K), c0);
    c1 = wmma_bf16(af, load_b_frag(W, N, kc * 32, n0 + 16, K), c1);
  }
  int lane = threadIdx.x & 31;
  int l = lane & 15, hi = lane >> 4;
#pragma unroll
  for (int g = 0; g < 8; ++g) {
    int m = m0 + g + hi * 8;
#pragma unroll
    for (int half = 0; half < 2; ++half) {
      int n = n0 + half * 16 + l;
      float v = (half ? c1[g] : c0[g]) + bias[n];
      if (do_tanh) v = tanhf(v);
      if (rowscale) v *= rowscale[m];
      size_t idx = (size_t)m * N + n;
      if (residual) v += residual[idx];
      out[idx] = v;
    }
  }
}

// ---- LU with partial pivoting, whole 256x256 matrix resident in LDS --------
// 1024 threads = 32 waves. Row stride padded to 257 floats so the column-k
// pivot scan (row-strided) rotates across the 64 LDS banks. Wave 0 does the
// pivot search (shfl reduction) and the row swap alone; trailing update needs
// no stored L (only log|pivot|), so each step is just 2 barriers.
__global__ __launch_bounds__(1024) void lu_logdet_kernel(const float* __restrict__ mats,
                                                         float* __restrict__ ld) {
  __shared__ float As[256 * 257];
  __shared__ float s_inv;
  const float* A = mats + (size_t)blockIdx.x * 256 * 256;
  int t = threadIdx.x;
  for (int idx = t; idx < 256 * 256; idx += 1024) {
    int i = idx >> 8, j = idx & 255;
    As[i * 257 + j] = A[idx];
  }
  __syncthreads();
  int lane = t & 31, wave = t >> 5;
  float logacc = 0.f;  // lives in thread 0's registers
  for (int k = 0; k < 256; ++k) {
    if (wave == 0) {
      // partial-pivot search over column k, rows k..255
      float mv = -1.f; int mi = k;
      for (int i = k + lane; i < 256; i += 32) {
        float v = fabsf(As[i * 257 + k]);
        if (v > mv) { mv = v; mi = i; }
      }
#pragma unroll
      for (int s = 16; s > 0; s >>= 1) {
        float ov = __shfl_xor(mv, s, 32);
        int   oi = __shfl_xor(mi, s, 32);
        if (ov > mv) { mv = ov; mi = oi; }
      }
      int piv = __shfl(mi, 0, 32);  // consistent pivot across lanes
      if (piv != k) {
        for (int j = lane; j < 256; j += 32) {
          float tmp = As[k * 257 + j];
          As[k * 257 + j] = As[piv * 257 + j];
          As[piv * 257 + j] = tmp;
        }
      }
      if (lane == 0) {
        float pv = As[k * 257 + k];
        logacc += logf(fabsf(pv));
        s_inv = 1.f / pv;
      }
    }
    __syncthreads();
    float inv = s_inv;
    // trailing update rows i>k (wave-strided), cols j>k (lane-strided).
    // multiplier computed on the fly; column k left untouched (L not needed).
    for (int i = k + 1 + wave; i < 256; i += 32) {
      float lf = As[i * 257 + k] * inv;
      for (int j = k + 1 + lane; j < 256; j += 32) {
        As[i * 257 + j] -= lf * As[k * 257 + j];
      }
    }
    __syncthreads();
  }
  if (t == 0) ld[blockIdx.x] = logacc;
}

__global__ void finalize_kernel(const float* __restrict__ ld, float* __restrict__ out) {
  out[0] = ld[0] + ld[1];  // sign cancels: log|psi| = log|det_u| + log|det_d|
}

extern "C" void kernel_launch(void* const* d_in, const int* in_sizes, int n_in,
                              void* d_out, int out_size, void* d_ws, size_t ws_size,
                              hipStream_t stream) {
  const float* r     = (const float*)d_in[0];
  const float* a     = (const float*)d_in[1];
  const float* V0_W  = (const float*)d_in[2];
  const float* V0_b  = (const float*)d_in[3];
  const float* Vr_W  = (const float*)d_in[4];   // (3,896,256)
  const float* Vr_b  = (const float*)d_in[5];   // (3,256)
  const float* W0_W  = (const float*)d_in[6];   // (4,64)
  const float* W0_b  = (const float*)d_in[7];
  const float* Wr_W  = (const float*)d_in[8];   // (3,64,64)
  const float* Wr_b  = (const float*)d_in[9];
  const float* Vhu_W = (const float*)d_in[10];
  const float* Vhu_b = (const float*)d_in[11];
  const float* Vhd_W = (const float*)d_in[12];
  const float* Vhd_b = (const float*)d_in[13];
  const float* wu_W  = (const float*)d_in[14];
  const float* wu_b  = (const float*)d_in[15];
  const float* wd_W  = (const float*)d_in[16];
  const float* wd_b  = (const float*)d_in[17];

  float* w = (float*)d_ws;
  size_t off = 0;
  float* P      = w + off; off += (size_t)NE * NE * 64;   // 64 MB
  float* s_v0   = w + off; off += (size_t)NE * 64;
  float* s_v    = w + off; off += (size_t)NE * 256;
  float* s_blk  = w + off; off += (size_t)NE * 896;
  float* su_m   = w + off; off += 256;
  float* sd_m   = w + off; off += 256;
  float* pu_m   = w + off; off += (size_t)NE * 64;
  float* pd_m   = w + off; off += (size_t)NE * 64;
  float* e_fac  = w + off; off += NE;
  float* head_u = w + off; off += 256 * 256;
  float* head_d = w + off; off += 256 * 256;
  float* orb    = w + off; off += 2 * 256 * 256;
  float* ldbuf  = w + off; off += 2;

  // features + envelope sums
  prep_kernel<<<2, 256, 0, stream>>>(r, a, s_v0, e_fac);
  smean_kernel<<<1, 64, 0, stream>>>(s_v0, 64, su_m, sd_m);
  p4mean_kernel<<<NE, 256, 0, stream>>>(r, pu_m, pd_m);
  // s_blk layer 0 (Din = 200)
  sblk_kernel<<<NE, 256, 0, stream>>>(s_v0, su_m, sd_m, pu_m, pd_m, 64, 4, s_blk);
  // layer 0: s_v = tanh(s_blk0 @ V0 + b), no residual
  sgemm_kernel<<<dim3(NE / 16, 256 / 32), 32, 0, stream>>>(
      s_blk, 200, V0_W, V0_b, s_v, 256, nullptr, nullptr, 1);
  // layer 0: P = tanh(p4 @ W0 + b)
  p0_kernel<<<NE * NE / 4, 256, 0, stream>>>(r, W0_W, W0_b, P);

  // residual layers 1..3
  for (int lyr = 1; lyr < 4; ++lyr) {
    pmean_kernel<<<NE, 64, 0, stream>>>(P, pu_m, pd_m);
    smean_kernel<<<1, 256, 0, stream>>>(s_v, 256, su_m, sd_m);
    sblk_kernel<<<NE, 256, 0, stream>>>(s_v, su_m, sd_m, pu_m, pd_m, 256, 64, s_blk);
    sgemm_kernel<<<dim3(NE / 16, 256 / 32), 32, 0, stream>>>(
        s_blk, 896, Vr_W + (size_t)(lyr - 1) * 896 * 256, Vr_b + (lyr - 1) * 256,
        s_v, 256, /*residual=*/s_v, nullptr, 1);
    pgemm_kernel<<<NE * NE / 128, 256, 0, stream>>>(
        P, Wr_W + (size_t)(lyr - 1) * 64 * 64, Wr_b + (lyr - 1) * 64);
  }

  // final fb_block
  pmean_kernel<<<NE, 64, 0, stream>>>(P, pu_m, pd_m);
  smean_kernel<<<1, 256, 0, stream>>>(s_v, 256, su_m, sd_m);
  sblk_kernel<<<NE, 256, 0, stream>>>(s_v, su_m, sd_m, pu_m, pd_m, 256, 64, s_blk);

  // heads: s_u = tanh(s_blk[:256] @ Vhu + b), s_d = tanh(s_blk[256:] @ Vhd + b)
  sgemm_kernel<<<dim3(NU / 16, 256 / 32), 32, 0, stream>>>(
      s_blk, 896, Vhu_W, Vhu_b, head_u, 256, nullptr, nullptr, 1);
  sgemm_kernel<<<dim3(NU / 16, 256 / 32), 32, 0, stream>>>(
      s_blk + (size_t)NU * 896, 896, Vhd_W, Vhd_b, head_d, 256, nullptr, nullptr, 1);

  // orbitals: orb = (head @ w + b) * envelope_rowscale  (no tanh)
  sgemm_kernel<<<dim3(NU / 16, 256 / 32), 32, 0, stream>>>(
      head_u, 256, wu_W, wu_b, orb, 256, nullptr, e_fac, 0);
  sgemm_kernel<<<dim3(NU / 16, 256 / 32), 32, 0, stream>>>(
      head_d, 256, wd_W, wd_b, orb + 256 * 256, 256, nullptr, e_fac + NU, 0);

  // log|det| of both 256x256 orbital matrices (LDS-resident LU), then combine
  lu_logdet_kernel<<<2, 1024, 0, stream>>>(orb, ldbuf);
  finalize_kernel<<<1, 1, 0, stream>>>(ldbuf, (float*)d_out);
}
